// conv3d_84971632984716
// MI455X (gfx1250) — matrix operand order repaired and checked by reading, not compiled, not run on hardware
//
#include <hip/hip_runtime.h>

typedef _Float16 v4h  __attribute__((ext_vector_type(4)));
typedef _Float16 v8h  __attribute__((ext_vector_type(8)));
typedef _Float16 v16h __attribute__((ext_vector_type(16)));
typedef float    v8f  __attribute__((ext_vector_type(8)));
typedef unsigned int u32x4 __attribute__((ext_vector_type(4)));
typedef int          i32x4 __attribute__((ext_vector_type(4)));
typedef int          i32x8 __attribute__((ext_vector_type(8)));

#define CIN  128
#define COUT 128
#define KOFF 27
#define LDS_STRIDE 136   // f16 per LDS row: 128 + 8 pad -> 272B rows, 16B aligned, bank-spread
#define BUF_ELEMS (CIN * LDS_STRIDE)         // f16 elements per weight buffer
#define BUF_BYTES (BUF_ELEMS * 2)            // 34816 B
#define LDS_TOTAL_BYTES (2 * BUF_BYTES)      // 69632 B (double buffer, dynamic LDS)

// fp32 -> fp16 conversion; region [n_src, n_tot) is written as zeros
// (used to append a zero feature row for masked / out-of-range neighbors).
__global__ __launch_bounds__(256) void cvt_f32_to_f16(const float* __restrict__ src,
                                                      _Float16* __restrict__ dst,
                                                      long n_src, long n_tot) {
  long i = ((long)blockIdx.x * blockDim.x + threadIdx.x) * 4;
  if (i >= n_tot) return;
  float4 v = make_float4(0.f, 0.f, 0.f, 0.f);
  if (i < n_src) v = *(const float4*)(src + i);
  v4h h;
  h[0] = (_Float16)v.x; h[1] = (_Float16)v.y;
  h[2] = (_Float16)v.z; h[3] = (_Float16)v.w;
  *(v4h*)(dst + i) = h;
}

// Issue one TDM DMA: W[k] (128x128 f16, contiguous 32KB) -> LDS with hardware
// padding of 4 dwords after every 64 dwords (row stride 128 -> 136 f16).
// D# layout per CDNA5 ISA. Uniform args -> SGPR descriptors.
// This toolchain exposes the 6-arg builtin: (g0, g1, g2, g3, extra, cpol);
// groups 2/3 and the extra group are zero-filled (2D/1D tile, no gather/iterate).
__device__ __forceinline__ void issue_weight_tdm(const _Float16* gsrc, unsigned lds_off) {
  unsigned long long ga = (unsigned long long)(uintptr_t)gsrc;
  u32x4 g0;
  g0[0] = 1u;                                   // count=1, user mode, no gather
  g0[1] = lds_off;                              // lds_addr (bytes)
  g0[2] = (unsigned)ga;                         // global_addr[31:0]
  g0[3] = (unsigned)((ga >> 32) & 0x1FFFFFFu)   // global_addr[56:32]
        | (2u << 30);                           // type = 2 ("image")
  i32x8 g1;
  g1[0] = (1 << 16)                             // data_size = 1 -> 2 bytes
        | (1 << 20)                             // pad_enable
        | (5 << 22)                             // pad_interval: code 5 -> every 64 dwords
        | (3 << 25);                            // pad_amount:  code 3 -> 4 dwords (8 f16)
  g1[1] = (int)(16384u << 16);                  // tensor_dim0 = 16384 (low 16 @ bits 63:48)
  g1[2] = (1 << 16);                            // tensor_dim0 hi=0; tensor_dim1 = 1
  g1[3] = (int)(16384u << 16);                  // tensor_dim1 hi=0; tile_dim0 = 16384
  g1[4] = 0;                                    // tile_dim1 = 0 (unused), tile_dim2 = 0
  g1[5] = 16384;                                // tensor_dim0_stride low 32
  g1[6] = 0;                                    // stride hi; tensor_dim1_stride lo
  g1[7] = 0;
  i32x4 z4; z4[0] = 0; z4[1] = 0; z4[2] = 0; z4[3] = 0;
  i32x8 z8;
  #pragma unroll
  for (int i = 0; i < 8; ++i) z8[i] = 0;
  __builtin_amdgcn_tensor_load_to_lds(g0, g1, z4, z4, z8, 0);
}

// Submanifold sparse conv via WMMA.
// Wave: 16 voxel rows x 128 Cout (8x v_wmma_f32_16x16x32_f16 tiles, f32 accum).
// Block: 8 waves = 128 voxel rows. W[k] staged by the Tensor Data Mover into a
// double-buffered padded LDS image, overlapped with compute on the other buffer.
__global__ __launch_bounds__(256) void spconv_wmma_f16(
    const _Float16* __restrict__ featsH,   // (N+1) x 128 f16, row N = zeros
    const _Float16* __restrict__ wH,       // 27 x 128 x 128 f16
    const int*      __restrict__ nbr,      // 27 x N
    float*          __restrict__ out,      // N x 128 f32
    int nvox)
{
  extern __shared__ _Float16 sw[];         // 2 x BUF_ELEMS (dynamic: 69632 B)

  const int tid  = threadIdx.x;
  const int lane = tid & 31;
  const int wave = tid >> 5;

  const int  tileBase   = ((int)blockIdx.x * 8 + wave) * 16;
  const bool activeWave = (tileBase < nvox);          // wave-uniform

  const int mrow   = lane & 15;     // M index of this lane
  const int hiHalf = lane >> 4;     // 0: lanes 0-15, 1: lanes 16-31
  const int gRow   = tileBase + mrow;

  const unsigned ldsBase = (unsigned)(uintptr_t)(&sw[0]);  // LDS byte offset

  v8f acc[8];
  #pragma unroll
  for (int t = 0; t < 8; ++t) acc[t] = (v8f)(0.0f);

  // Prologue: DMA W[0] into buffer 0 (one wave issues; TENSORcnt tracks it).
  if (wave == 0) issue_weight_tdm(wH, ldsBase);

  for (int k = 0; k < KOFF; ++k) {
    const int buf = k & 1;
    if (wave == 0) __builtin_amdgcn_s_wait_tensorcnt(0);
    __syncthreads();   // W[k] visible to all waves; prior readers of buf^1 done
    if (wave == 0 && (k + 1) < KOFF)
      issue_weight_tdm(wH + (size_t)(k + 1) * CIN * COUT,
                       ldsBase + (unsigned)(((k + 1) & 1) * BUF_BYTES));

    if (activeWave) {   // wave-uniform: EXEC stays all-ones for WMMA
      int idx = (gRow < nvox) ? nbr[(size_t)k * nvox + gRow] : -1;
      // branchless masked gather: invalid neighbors read the zero row (row N)
      const _Float16* rowp = featsH + (size_t)(idx < 0 ? nvox : idx) * CIN;
      const _Float16* wbuf = sw + buf * BUF_ELEMS;

      #pragma unroll
      for (int cb = 0; cb < 4; ++cb) {          // Cin blocks of 32 (WMMA K)
        // A frag, 16-bit 16x32 layout: lanes 0-15 K=0-7/16-23, lanes 16-31 K=8-15/24-31
        const int c0 = cb * 32 + hiHalf * 8;
        v8h alo = *(const v8h*)(rowp + c0);
        v8h ahi = *(const v8h*)(rowp + c0 + 16);
        v16h a = __builtin_shufflevector(alo, ahi,
                 0,1,2,3,4,5,6,7,8,9,10,11,12,13,14,15);

        // B frag: N = lane%16, K same order as A (8h+s / 16+8h+s-8)
        const _Float16* bbase = wbuf + (cb * 32 + hiHalf * 8) * LDS_STRIDE + (lane & 15);
        #pragma unroll
        for (int nt = 0; nt < 8; ++nt) {        // Cout tiles of 16
          v16h b;
          #pragma unroll
          for (int s = 0; s < 8; ++s) {
            b[s]     = bbase[s * LDS_STRIDE + nt * 16];
            b[s + 8] = bbase[(s + 16) * LDS_STRIDE + nt * 16];
          }
          acc[nt] = __builtin_amdgcn_wmma_f32_16x16x32_f16(
              /*neg_a=*/false, a, /*neg_b=*/false, b,
              /*c_mod=*/(short)0, acc[nt],
              /*reuse_a=*/false, /*reuse_b=*/false);
        }
      }
    }
  }

  if (activeWave) {
    // C/D layout: VGPR r -> M = r + 8*hiHalf, N = lane%16
    const int col0 = lane & 15;
    #pragma unroll
    for (int nt = 0; nt < 8; ++nt) {
      #pragma unroll
      for (int r = 0; r < 8; ++r) {
        int row = tileBase + hiHalf * 8 + r;
        if (row < nvox)
          out[(size_t)row * COUT + nt * 16 + col0] = acc[nt][r];
      }
    }
  }
}

extern "C" void kernel_launch(void* const* d_in, const int* in_sizes, int n_in,
                              void* d_out, int out_size, void* d_ws, size_t ws_size,
                              hipStream_t stream) {
  const float* feats = (const float*)d_in[0];  // [N,128] f32
  const float* kern  = (const float*)d_in[1];  // [27,128,128] f32
  const int*   nbr   = (const int*)d_in[2];    // [27,N] i32
  float* out = (float*)d_out;                  // [N,128] f32

  const int N = in_sizes[0] / CIN;

  // Workspace: f16 feats with appended zero row, then f16 weights (~13.7 MB total)
  _Float16* featsH = (_Float16*)d_ws;
  _Float16* wH     = featsH + (size_t)(N + 1) * CIN;   // 256B-aligned offset

  {
    long n_src = (long)N * CIN;
    long n_tot = (long)(N + 1) * CIN;
    long thr   = (n_tot + 3) / 4;
    int  blk   = (int)((thr + 255) / 256);
    cvt_f32_to_f16<<<blk, 256, 0, stream>>>(feats, featsH, n_src, n_tot);
  }
  {
    long n   = (long)KOFF * CIN * COUT;
    long thr = (n + 3) / 4;
    int  blk = (int)((thr + 255) / 256);
    cvt_f32_to_f16<<<blk, 256, 0, stream>>>(kern, wH, n, n);
  }

  int tiles  = (N + 15) / 16;
  int blocks = (tiles + 7) / 8;
  spconv_wmma_f16<<<blocks, 256, LDS_TOTAL_BYTES, stream>>>(featsH, wH, nbr, out, N);
}
